// NMPEncoder_79517024518960
// MI455X (gfx1250) — matrix operand order repaired and checked by reading, not compiled, not run on hardware
//
#include <hip/hip_runtime.h>
#include <hip/hip_bf16.h>
#include <hip/hip_fp16.h>

// ---------------- WMMA types (CDNA5 / gfx1250, wave32) ----------------
typedef _Float16 v16h __attribute__((ext_vector_type(16)));
typedef float    v8f  __attribute__((ext_vector_type(8)));

#define HD __device__ __forceinline__

HD v8f wmma_f16(v16h a, v16h b, v8f c) {
  // D = A(16x32 f16) * B(32x16 f16) + C(16x16 f32)
  return __builtin_amdgcn_wmma_f32_16x16x32_f16(
      /*neg_a=*/false, a, /*neg_b=*/false, b,
      /*c_mod=*/(short)0, c, /*reuse_a=*/false, /*reuse_b=*/false);
}

// Branchless ELU using the hardware transcendental (v_exp_f32, TRANS pipe):
//   x > 0 : max(x,0)=x, exp(min(x,0))-1 = 0      -> x
//   x <= 0: max(x,0)=0, exp(x)-1                 -> expm1-ish (1-ulp exp error, fine)
HD float eluf(float x) {
  return fmaxf(x, 0.f) + (__expf(fminf(x, 0.f)) - 1.f);
}

// LDS tile strides padded by 8 halves (16B) so consecutive rows shift 4 banks:
// unpadded 512/256-half strides map every row to the same bank (16-way conflict
// on the 16-lane A-operand ds_load_b128); padded strides spread across 64 banks.
constexpr int LDA = 512 + 8;   // A tile row stride (halves)
constexpr int LDH = 256 + 8;   // h1/fused tile row stride (halves)

// A operand (16x32 f16 tile) from row-major LDS.
// ISA layout: lanes 0-15 -> M=lane, halves 0..7 = K[kbase+8*hi .. +7],
//             halves 8..15 = K[kbase+16+8*hi .. +7]; hi = lane>>4.
HD v16h load_a_lds(const _Float16* lds, int rowStride, int m, int hi, int kbase) {
  v16h a;
  const _Float16* p0 = lds + m * rowStride + kbase + hi * 8;
#pragma unroll
  for (int j = 0; j < 8; ++j) { a[j] = p0[j]; a[8 + j] = p0[16 + j]; }
  return a;
}

// B operand: pre-packed so each lane's 16 halves are contiguous (32B).
HD v16h load_b_packed(const _Float16* p) {
  v16h b;
#pragma unroll
  for (int j = 0; j < 16; ++j) b[j] = p[j];
  return b;
}

// ---------------- problem constants ----------------
constexpr int Bn = 16, Nn = 64, Hh = 256, En = 4032, PREDn = 30;
constexpr int ROWS_E = Bn * En;        // 64512
constexpr int ROWS_N = Bn * Nn;        // 1024
constexpr int TILES_PER_BATCH = En / 16;  // 252

// ---------------- workspace layout (bytes) ----------------
constexpr size_t OFF_XH    = 0;                                        // f16 [1024][256]
constexpr size_t OFF_H2E   = OFF_XH    + (size_t)ROWS_N * Hh * 2;      // f16 [64512][256]
constexpr size_t OFF_NODES = OFF_H2E   + (size_t)ROWS_E * Hh * 2;      // f16 [1024][512]
constexpr size_t OFF_H2M   = OFF_NODES + (size_t)ROWS_N * 512 * 2;     // f16 [1024][256]
constexpr size_t OFF_W1E   = OFF_H2M   + (size_t)ROWS_N * Hh * 2;      // f16 packed 512x256
constexpr size_t OFF_W2E   = OFF_W1E   + (size_t)131072 * 2;           // f16 packed 256x256
constexpr size_t OFF_W1M   = OFF_W2E   + (size_t)65536 * 2;
constexpr size_t OFF_W2M   = OFF_W1M   + (size_t)131072 * 2;
constexpr size_t OFF_WF    = OFF_W2M   + (size_t)65536 * 2;
constexpr size_t OFF_WP    = OFF_WF    + (size_t)131072 * 2;           // f16 packed 256x64
constexpr size_t OFF_ST    = OFF_WP    + (size_t)16384 * 2;            // f32 stats [8][256]

// ---------------- small kernels ----------------
__global__ void k_zero(float* p, int n) {
  int i = blockIdx.x * 256 + threadIdx.x;
  if (i < n) p[i] = 0.f;
}

// Pack row-major W[K][realN] (f32) into WMMA B layout:
// dst[((ct*KT + kt)*32 + lane)*16 + h] = W[kt*32 + (h>>3)*16 + (lane>>4)*8 + (h&7)][ct*16 + (lane&15)]
__global__ void k_pack(const float* __restrict__ W, _Float16* __restrict__ dst,
                       int K, int realN, int NCpad) {
  int p = blockIdx.x * 256 + threadIdx.x;
  int total = K * NCpad;
  if (p >= total) return;
  int h    = p & 15;
  int lane = (p >> 4) & 31;
  int tile = p >> 9;
  int KT = K >> 5;
  int kt = tile % KT;
  int ct = tile / KT;
  int k = kt * 32 + (h >> 3) * 16 + (lane >> 4) * 8 + (h & 7);
  int n = ct * 16 + (lane & 15);
  float v = (n < realN) ? W[(size_t)k * realN + n] : 0.f;
  dst[p] = (_Float16)v;
}

// traj embed: x = rel_pos(12) @ W_traj(12x256) + b_traj; store f16
__global__ void k_traj(const float* __restrict__ prev, const float* __restrict__ Wt,
                       const float* __restrict__ bt, _Float16* __restrict__ xh) {
  int g = blockIdx.x;            // node-row 0..1023
  int col = threadIdx.x;
  const float* p = prev + (size_t)g * 12;   // (PREV=6, 2)
  float acc = bt[col];
#pragma unroll
  for (int t = 1; t < 6; ++t)
#pragma unroll
    for (int c = 0; c < 2; ++c) {
      float d = p[t * 2 + c] - p[(t - 1) * 2 + c];
      acc += d * Wt[(size_t)(t * 2 + c) * 256 + col];
    }
  xh[(size_t)g * 256 + col] = (_Float16)acc;
}

__global__ void k_bnfin(const float* __restrict__ sum, const float* __restrict__ sq,
                        const float* __restrict__ gamma, const float* __restrict__ beta,
                        float* __restrict__ s, float* __restrict__ t, float inv_cnt) {
  int h = threadIdx.x;
  float mu  = sum[h] * inv_cnt;
  float var = sq[h] * inv_cnt - mu * mu;
  float rs  = rsqrtf(var + 1e-5f);
  float sc  = rs * gamma[h];
  s[h] = sc;
  t[h] = beta[h] - mu * sc;
}

// ---------------- fused 2-layer MLP tile (512->256 ELU, 256->256 ELU) ----------------
// sA: staged 16x512 f16 A tile (stride LDA). Emits h2 (f16) + f32 stat atomics.
HD void mlp2_tile(const _Float16* sA, _Float16* sH,
                  const _Float16* __restrict__ W1p, const float* __restrict__ b1,
                  const _Float16* __restrict__ W2p, const float* __restrict__ b2,
                  _Float16* __restrict__ h2out,   // tile base, row stride 256
                  float* __restrict__ sumv, float* __restrict__ sumsq) {
  int tid = threadIdx.x, lane = tid & 31, wid = tid >> 5;
  int m = lane & 15, hi = lane >> 4;
  int col0 = (wid * 2 + 0) * 16 + m;
  int col1 = (wid * 2 + 1) * 16 + m;

  // Layer 1: K = 512 (16 k-tiles)
  v8f a0, a1;
  {
    float bv0 = b1[col0], bv1 = b1[col1];
#pragma unroll
    for (int r = 0; r < 8; ++r) { a0[r] = bv0; a1[r] = bv1; }
  }
  for (int kt = 0; kt < 16; ++kt) {
    v16h av = load_a_lds(sA, LDA, m, hi, kt * 32);
    v16h b0 = load_b_packed(W1p + ((size_t)((wid * 2 + 0) * 16 + kt) * 32 + lane) * 16);
    v16h b1v = load_b_packed(W1p + ((size_t)((wid * 2 + 1) * 16 + kt) * 32 + lane) * 16);
    a0 = wmma_f16(av, b0, a0);
    a1 = wmma_f16(av, b1v, a1);
  }
  // ELU -> stage h1 in LDS (f16, row-major 16 x LDH)
#pragma unroll
  for (int r = 0; r < 8; ++r) {
    int M = r + hi * 8;
    sH[M * LDH + col0] = (_Float16)eluf(a0[r]);
    sH[M * LDH + col1] = (_Float16)eluf(a1[r]);
  }
  __syncthreads();

  // Layer 2: K = 256 (8 k-tiles)
  v8f c0, c1;
  {
    float bv0 = b2[col0], bv1 = b2[col1];
#pragma unroll
    for (int r = 0; r < 8; ++r) { c0[r] = bv0; c1[r] = bv1; }
  }
  for (int kt = 0; kt < 8; ++kt) {
    v16h av = load_a_lds(sH, LDH, m, hi, kt * 32);
    v16h b0 = load_b_packed(W2p + ((size_t)((wid * 2 + 0) * 8 + kt) * 32 + lane) * 16);
    v16h b1v = load_b_packed(W2p + ((size_t)((wid * 2 + 1) * 8 + kt) * 32 + lane) * 16);
    c0 = wmma_f16(av, b0, c0);
    c1 = wmma_f16(av, b1v, c1);
  }
  // ELU, store f16, accumulate BN stats (linear-BN trick downstream)
  float s0 = 0.f, q0 = 0.f, s1 = 0.f, q1 = 0.f;
#pragma unroll
  for (int r = 0; r < 8; ++r) {
    int M = r + hi * 8;
    float v0 = eluf(c0[r]);
    float v1 = eluf(c1[r]);
    h2out[(size_t)M * 256 + col0] = (_Float16)v0;
    h2out[(size_t)M * 256 + col1] = (_Float16)v1;
    s0 += v0; q0 += v0 * v0; s1 += v1; q1 += v1 * v1;
  }
  atomicAdd(sumv + col0, s0);  atomicAdd(sumsq + col0, q0);
  atomicAdd(sumv + col1, s1);  atomicAdd(sumsq + col1, q1);
}

// ---------------- edge MLP: gather(recv,send) fused into A staging ----------------
__global__ __launch_bounds__(256) void k_edge_mlp(
    const _Float16* __restrict__ xh,
    const _Float16* __restrict__ W1p, const float* __restrict__ b1,
    const _Float16* __restrict__ W2p, const float* __restrict__ b2,
    _Float16* __restrict__ h2e, float* __restrict__ sumv, float* __restrict__ sumsq) {
  __shared__ alignas(16) _Float16 sA[16 * LDA];
  __shared__ alignas(16) _Float16 sH[16 * LDH];
  int tid = threadIdx.x;
  int blk = blockIdx.x;
  int b = blk / TILES_PER_BATCH;
  int t = blk - b * TILES_PER_BATCH;
  int row0 = t * 16;                         // edge index base within batch
  {
    int row = tid >> 4, seg = tid & 15;      // seg: 16 chunks of 32 halves
    int e = row0 + row;
    int i = e / 63;
    int rr = e - i * 63;
    int j = rr + (rr >= i ? 1 : 0);          // recv = i, send = j
    const _Float16* src = (seg < 8)
        ? xh + ((size_t)(b * 64 + i) * 256 + seg * 32)
        : xh + ((size_t)(b * 64 + j) * 256 + (seg - 8) * 32);
    uint4* d4 = (uint4*)(sA + row * LDA + seg * 32);
    const uint4* s4 = (const uint4*)src;
#pragma unroll
    for (int q = 0; q < 4; ++q) d4[q] = s4[q];
  }
  __syncthreads();
  mlp2_tile(sA, sH, W1p, b1, W2p, b2,
            h2e + (size_t)(b * En + row0) * 256, sumv, sumsq);
}

// ---------------- aggregation: segment sums + linear BN affine ----------------
__global__ void k_aggregate(const _Float16* __restrict__ h2e,
                            const float* __restrict__ sE, const float* __restrict__ tE,
                            _Float16* __restrict__ nodesh) {
  int g = blockIdx.x;       // b*64 + n
  int b = g >> 6, n = g & 63;
  int h = threadIdx.x;
  const _Float16* base = h2e + (size_t)b * En * 256;
  float sin = 0.f, sout = 0.f;
  // incoming: recv==n -> contiguous rows [n*63, n*63+62]
  for (int k = 0; k < 63; ++k)
    sin += (float)base[(size_t)(n * 63 + k) * 256 + h];
  // outgoing: send==n -> e = i*63 + (n - (n>i))
  for (int i = 0; i < 64; ++i) {
    if (i == n) continue;
    int e = i * 63 + n - (n > i ? 1 : 0);
    sout += (float)base[(size_t)e * 256 + h];
  }
  float sc = sE[h], tc = tE[h];
  float inc  = sc * sin  * (1.f / 63.f) + tc;   // (s*Σv + 63*t)/63
  float outg = sc * sout * (1.f / 63.f) + tc;
  nodesh[(size_t)g * 512 + h]       = (_Float16)inc;
  nodesh[(size_t)g * 512 + 256 + h] = (_Float16)outg;
}

// ---------------- node MLP ----------------
__global__ __launch_bounds__(256) void k_node_mlp(
    const _Float16* __restrict__ nodesh,
    const _Float16* __restrict__ W1p, const float* __restrict__ b1,
    const _Float16* __restrict__ W2p, const float* __restrict__ b2,
    _Float16* __restrict__ h2m, float* __restrict__ sumv, float* __restrict__ sumsq) {
  __shared__ alignas(16) _Float16 sA[16 * LDA];
  __shared__ alignas(16) _Float16 sH[16 * LDH];
  int tid = threadIdx.x;
  int row0 = blockIdx.x * 16;
  {
    int row = tid >> 4, seg = tid & 15;
    const uint4* s4 = (const uint4*)(nodesh + (size_t)(row0 + row) * 512 + seg * 32);
    uint4* d4 = (uint4*)(sA + row * LDA + seg * 32);
#pragma unroll
    for (int q = 0; q < 4; ++q) d4[q] = s4[q];
  }
  __syncthreads();
  mlp2_tile(sA, sH, W1p, b1, W2p, b2, h2m + (size_t)row0 * 256, sumv, sumsq);
}

// ---------------- head: fuse GEMM + pred GEMM + cumsum ----------------
__global__ __launch_bounds__(256) void k_head(
    const _Float16* __restrict__ xh, const _Float16* __restrict__ h2m,
    const float* __restrict__ sM, const float* __restrict__ tM,
    const _Float16* __restrict__ WFp, const float* __restrict__ bF,
    const _Float16* __restrict__ WPp, const float* __restrict__ bP,
    const float* __restrict__ prev, float* __restrict__ out) {
  __shared__ alignas(16) _Float16 sA[16 * LDA];
  __shared__ alignas(16) _Float16 sF[16 * LDH];
  __shared__ alignas(16) float    sR[16 * 64];
  int tid = threadIdx.x, lane = tid & 31, wid = tid >> 5;
  int row0 = blockIdx.x * 16;
  {
    int row = tid >> 4, seg = tid & 15;
    int g = row0 + row;
    if (seg < 8) {
      const uint4* s4 = (const uint4*)(xh + (size_t)g * 256 + seg * 32);
      uint4* d4 = (uint4*)(sA + row * LDA + seg * 32);
#pragma unroll
      for (int q = 0; q < 4; ++q) d4[q] = s4[q];
    } else {
      int base = (seg - 8) * 32;
#pragma unroll
      for (int k = 0; k < 32; ++k) {
        int h = base + k;
        float v = sM[h] * (float)h2m[(size_t)g * 256 + h] + tM[h];  // BN affine on m
        sA[row * LDA + 256 + h] = (_Float16)v;
      }
    }
  }
  __syncthreads();

  int m = lane & 15, hi = lane >> 4;
  // fuse GEMM: [16x512] @ [512x256] + bF (no activation)
  {
    v8f a0, a1;
    int col0 = (wid * 2 + 0) * 16 + m;
    int col1 = (wid * 2 + 1) * 16 + m;
    float bv0 = bF[col0], bv1 = bF[col1];
#pragma unroll
    for (int r = 0; r < 8; ++r) { a0[r] = bv0; a1[r] = bv1; }
    for (int kt = 0; kt < 16; ++kt) {
      v16h av = load_a_lds(sA, LDA, m, hi, kt * 32);
      v16h b0 = load_b_packed(WFp + ((size_t)((wid * 2 + 0) * 16 + kt) * 32 + lane) * 16);
      v16h b1v = load_b_packed(WFp + ((size_t)((wid * 2 + 1) * 16 + kt) * 32 + lane) * 16);
      a0 = wmma_f16(av, b0, a0);
      a1 = wmma_f16(av, b1v, a1);
    }
#pragma unroll
    for (int r = 0; r < 8; ++r) {
      int M = r + hi * 8;
      sF[M * LDH + col0] = (_Float16)a0[r];
      sF[M * LDH + col1] = (_Float16)a1[r];
    }
  }
  __syncthreads();

  // pred GEMM: [16x256] @ [256x64(pad)] + bP ; waves 0..3 own the 4 col-tiles
  if (wid < 4) {
    int cti = wid;
    int col = cti * 16 + m;
    v8f c;
    float bv = (col < 60) ? bP[col] : 0.f;
#pragma unroll
    for (int r = 0; r < 8; ++r) c[r] = bv;
    for (int kt = 0; kt < 8; ++kt) {
      v16h av = load_a_lds(sF, LDH, m, hi, kt * 32);
      v16h bm = load_b_packed(WPp + ((size_t)(cti * 8 + kt) * 32 + lane) * 16);
      c = wmma_f16(av, bm, c);
    }
#pragma unroll
    for (int r = 0; r < 8; ++r) sR[(r + hi * 8) * 64 + col] = c[r];
  }
  __syncthreads();

  // rel out + cumsum -> pos out (2 threads per row: x/y component)
  if (tid < 32) {
    int row = tid >> 1, comp = tid & 1;
    int g = row0 + row;
    float run = prev[((size_t)g * 6 + 5) * 2 + comp];  // last observed location
    for (int k = 0; k < PREDn; ++k) {
      float v = sR[row * 64 + 2 * k + comp];
      out[(size_t)g * 60 + 2 * k + comp] = v;                       // rel
      run += v;
      out[(size_t)ROWS_N * 60 + (size_t)g * 60 + 2 * k + comp] = run;  // pos
    }
  }
}

// ---------------- host launcher ----------------
extern "C" void kernel_launch(void* const* d_in, const int* in_sizes, int n_in,
                              void* d_out, int out_size, void* d_ws, size_t ws_size,
                              hipStream_t stream) {
  const float* prev    = (const float*)d_in[0];
  const float* W_traj  = (const float*)d_in[3];
  const float* b_traj  = (const float*)d_in[4];
  const float* n2e_W1  = (const float*)d_in[5];
  const float* n2e_b1  = (const float*)d_in[6];
  const float* n2e_W2  = (const float*)d_in[7];
  const float* n2e_b2  = (const float*)d_in[8];
  const float* n2e_g   = (const float*)d_in[9];
  const float* n2e_be  = (const float*)d_in[10];
  const float* e2n_W1  = (const float*)d_in[11];
  const float* e2n_b1  = (const float*)d_in[12];
  const float* e2n_W2  = (const float*)d_in[13];
  const float* e2n_b2  = (const float*)d_in[14];
  const float* e2n_g   = (const float*)d_in[15];
  const float* e2n_be  = (const float*)d_in[16];
  const float* W_fuse  = (const float*)d_in[17];
  const float* b_fuse  = (const float*)d_in[18];
  const float* W_pred  = (const float*)d_in[19];
  const float* b_pred  = (const float*)d_in[20];

  char* ws = (char*)d_ws;
  _Float16* xh     = (_Float16*)(ws + OFF_XH);
  _Float16* h2e    = (_Float16*)(ws + OFF_H2E);
  _Float16* nodesh = (_Float16*)(ws + OFF_NODES);
  _Float16* h2m    = (_Float16*)(ws + OFF_H2M);
  _Float16* W1Ep   = (_Float16*)(ws + OFF_W1E);
  _Float16* W2Ep   = (_Float16*)(ws + OFF_W2E);
  _Float16* W1Mp   = (_Float16*)(ws + OFF_W1M);
  _Float16* W2Mp   = (_Float16*)(ws + OFF_W2M);
  _Float16* WFp    = (_Float16*)(ws + OFF_WF);
  _Float16* WPp    = (_Float16*)(ws + OFF_WP);
  float* stats = (float*)(ws + OFF_ST);
  float* sumE = stats + 0;    float* sqE = stats + 256;
  float* sumM = stats + 512;  float* sqM = stats + 768;
  float* sE   = stats + 1024; float* tE  = stats + 1280;
  float* sM   = stats + 1536; float* tM  = stats + 1792;

  float* out = (float*)d_out;

  // 0) zero BN accumulators (must happen every call: atomics accumulate)
  k_zero<<<8, 256, 0, stream>>>(stats, 2048);

  // 1) pack all GEMM weights into WMMA B-operand layout (f16)
  k_pack<<<512, 256, 0, stream>>>(n2e_W1, W1Ep, 512, 256, 256);
  k_pack<<<256, 256, 0, stream>>>(n2e_W2, W2Ep, 256, 256, 256);
  k_pack<<<512, 256, 0, stream>>>(e2n_W1, W1Mp, 512, 256, 256);
  k_pack<<<256, 256, 0, stream>>>(e2n_W2, W2Mp, 256, 256, 256);
  k_pack<<<512, 256, 0, stream>>>(W_fuse, WFp, 512, 256, 256);
  k_pack<<<64,  256, 0, stream>>>(W_pred, WPp, 256, 60, 64);

  // 2) traj embedding (K=12 -> plain VALU)
  k_traj<<<ROWS_N, 256, 0, stream>>>(prev, W_traj, b_traj, xh);

  // 3) edge MLP (gather fused, WMMA, ELU, f32 BN-stat atomics)
  k_edge_mlp<<<Bn * TILES_PER_BATCH, 256, 0, stream>>>(
      xh, W1Ep, n2e_b1, W2Ep, n2e_b2, h2e, sumE, sqE);

  // 4) finalize edge BN affine
  k_bnfin<<<1, 256, 0, stream>>>(sumE, sqE, n2e_g, n2e_be, sE, tE, 1.f / (float)ROWS_E);

  // 5) aggregate edges -> nodes (segment sums + linear BN)
  k_aggregate<<<ROWS_N, 256, 0, stream>>>(h2e, sE, tE, nodesh);

  // 6) node MLP
  k_node_mlp<<<ROWS_N / 16, 256, 0, stream>>>(
      nodesh, W1Mp, e2n_b1, W2Mp, e2n_b2, h2m, sumM, sqM);

  // 7) finalize node BN affine
  k_bnfin<<<1, 256, 0, stream>>>(sumM, sqM, e2n_g, e2n_be, sM, tM, 1.f / (float)ROWS_N);

  // 8) fuse + pred + cumsum -> (rel, pos)
  k_head<<<ROWS_N / 16, 256, 0, stream>>>(
      xh, h2m, sM, tM, WFp, b_fuse, WPp, b_pred, prev, out);
}
